// EarlyExitWeightedModel_72980084293649
// MI455X (gfx1250) — matrix operand order repaired and checked by reading, not compiled, not run on hardware
//
#include <hip/hip_runtime.h>
#include <hip/hip_bf16.h>
#include <cstdint>

typedef __attribute__((ext_vector_type(16))) _Float16 v16h;
typedef __attribute__((ext_vector_type(8)))  _Float16 v8h;
typedef __attribute__((ext_vector_type(8)))  float    v8f;
typedef __attribute__((ext_vector_type(4)))  float    v4f;
typedef __attribute__((ext_vector_type(4)))  unsigned int v4u;
typedef __attribute__((ext_vector_type(8)))  int      v8i;
typedef __attribute__((ext_vector_type(4)))  int      v4i;

#define Bdim 64
#define Tdim 256
#define Ddim 768
#define NHdim 12
#define Cdim 7
#define DHdim 64
#define SCPITCH 260   // LDS score row pitch (floats); 260*4 % 16 == 0

__device__ __forceinline__ v8f zero8() {
    v8f z = {0.f, 0.f, 0.f, 0.f, 0.f, 0.f, 0.f, 0.f};
    return z;
}

__device__ __forceinline__ v8f wmma_f16(v16h a, v16h b, v8f c) {
    // D = A(16x32 f16) * B(32x16 f16) + C(16x16 f32)
    return __builtin_amdgcn_wmma_f32_16x16x32_f16(
        false, a, false, b, (short)0, c, false, false);
}

__device__ __forceinline__ v16h cat8(v8h lo, v8h hi) {
    return __builtin_shufflevector(lo, hi, 0, 1, 2, 3, 4, 5, 6, 7,
                                   8, 9, 10, 11, 12, 13, 14, 15);
}

// ---------------------------------------------------------------------------
// Fragment builders (wave32; contiguous K order, same for A and B).
// A (16x32): row M=lane%16; lane-half h holds K {16h..16h+15}
//            -> one contiguous 32B run   => 2x b128 loads.
// B (32x16): col N=lane%16; lane-half h holds K {16h..16h+15}
//            -> one contiguous 32B run   => 2x b128 loads (K-major source).
// ---------------------------------------------------------------------------
__device__ __forceinline__ v16h load_a_f16(const _Float16* __restrict__ A,
                                           int lda, size_t row0, int k0, int lane) {
    int half = (lane >> 4) & 1, m = lane & 15;
    const _Float16* p = A + (row0 + (size_t)m) * lda + k0 + half * 16;
    v8h lo = *(const v8h*)p;          // K: 16h .. 16h+7
    v8h hi = *(const v8h*)(p + 8);    // K: 16h+8 .. 16h+15
    return cat8(lo, hi);
}

__device__ __forceinline__ v16h load_a_f32(const float* __restrict__ A,
                                           int lda, size_t row0, int k0, int lane) {
    int half = (lane >> 4) & 1, m = lane & 15;
    const float* p = A + (row0 + (size_t)m) * lda + k0 + half * 16;
    v4f a0 = *(const v4f*)p;
    v4f a1 = *(const v4f*)(p + 4);
    v4f a2 = *(const v4f*)(p + 8);
    v4f a3 = *(const v4f*)(p + 12);
    v16h a;
#pragma unroll
    for (int i = 0; i < 4; ++i) {
        a[i]      = (_Float16)a0[i];
        a[i + 4]  = (_Float16)a1[i];
        a[i + 8]  = (_Float16)a2[i];
        a[i + 12] = (_Float16)a3[i];
    }
    return a;
}

// B from pre-packed transposed f16 weights Wt[N][K]: contiguous per lane.
__device__ __forceinline__ v16h load_b_f16t(const _Float16* __restrict__ Wt,
                                            int ldk, int k0, int col0, int lane) {
    int half = (lane >> 4) & 1, n = lane & 15;
    const _Float16* p = Wt + (size_t)(col0 + n) * ldk + k0 + half * 16;
    return cat8(*(const v8h*)p, *(const v8h*)(p + 8));
}

// ---------------------------------------------------------------------------
// Tensor Data Mover: stage a 2D tile (tile_w x tile_h elems, 2-byte elems)
// of a row-major tensor (row width/stride = tensor_w elems) into LDS.
// D# layout per CDNA5 ISA (group0 + group1).
// ---------------------------------------------------------------------------
__device__ __forceinline__ void tdm_load_2d_f16(unsigned lds_off,
                                                const _Float16* gptr,
                                                unsigned tensor_w, unsigned tensor_h,
                                                unsigned tile_w, unsigned tile_h) {
    unsigned long long ga = (unsigned long long)(uintptr_t)gptr;
    v4u g0;
    g0[0] = 1u;                                      // count=1, user mode
    g0[1] = lds_off;                                 // lds_addr (bytes)
    g0[2] = (unsigned)(ga & 0xffffffffu);            // global_addr[31:0]
    g0[3] = (unsigned)((ga >> 32) & 0x01ffffffu)     // global_addr[56:32]
          | (2u << 30);                              // type=2 ("image")
    v8i g1;
    g1[0] = (int)(1u << 16);                         // wg_mask=0, data_size=1 (2B)
    g1[1] = (int)((tensor_w & 0xffffu) << 16);       // tensor_dim0[15:0]
    g1[2] = (int)((tensor_w >> 16) & 0xffffu)        // tensor_dim0[31:16]
          | (int)((tensor_h & 0xffffu) << 16);       // tensor_dim1[15:0]
    g1[3] = (int)((tensor_h >> 16) & 0xffffu)        // tensor_dim1[31:16]
          | (int)((tile_w & 0xffffu) << 16);         // tile_dim0
    g1[4] = (int)(tile_h & 0xffffu);                 // tile_dim1; tile_dim2=0
    g1[5] = (int)tensor_w;                           // tensor_dim0_stride[31:0]
    g1[6] = 0;                                       // stride hi / dim1_stride lo
    g1[7] = 0;
    v4i z4 = {0, 0, 0, 0};
#if defined(__clang_major__) && (__clang_major__ >= 23)
    v8i z8 = {0, 0, 0, 0, 0, 0, 0, 0};
    __builtin_amdgcn_tensor_load_to_lds(g0, g1, z4, z4, z8, 0);
#else
    __builtin_amdgcn_tensor_load_to_lds(g0, g1, z4, z4, 0);
#endif
}

// ---------------------------------------------------------------------------
// Kernel 0: pack Wq/Wk/Wv/Wo ([K][N] fp32) -> transposed f16 [N][K].
// One-time ~9.4 MB read; makes every GEMM B-fragment two b128 loads.
// ---------------------------------------------------------------------------
__global__ void wpack_kernel(const float* __restrict__ Wq,
                             const float* __restrict__ Wk,
                             const float* __restrict__ Wv,
                             const float* __restrict__ Wo,
                             _Float16* __restrict__ wt) {
    size_t idx = (size_t)blockIdx.x * blockDim.x + threadIdx.x;
    const size_t E = (size_t)Ddim * Ddim;
    if (idx >= 4 * E) return;
    int which = (int)(idx / E);
    size_t e = idx % E;
    int nrow = (int)(e / Ddim);               // output row = column of W
    int k    = (int)(e % Ddim);
    const float* W = (which == 0) ? Wq : (which == 1) ? Wk
                   : (which == 2) ? Wv : Wo;
    wt[idx] = (_Float16)W[(size_t)k * Ddim + nrow];
}

// ---------------------------------------------------------------------------
// Kernel 1: fused Q/K/V projection.  grid=(1024,18), block=256 (8 waves).
// ---------------------------------------------------------------------------
__global__ void qkv_kernel(const float* __restrict__ H,
                           const _Float16* __restrict__ wt,   // packed Wq|Wk|Wv|Wo
                           const float* __restrict__ bq,
                           const float* __restrict__ bk,
                           const float* __restrict__ bv,
                           _Float16* __restrict__ qf, _Float16* __restrict__ kf,
                           _Float16* __restrict__ vf) {
    int lane = threadIdx.x & 31;
    int w    = threadIdx.x >> 5;
    int nt   = blockIdx.y * 8 + w;          // 0..143 = 3 outputs * 48 col-tiles
    int which = nt / 48;
    int ntile = nt % 48;
    const _Float16* Wt  = wt + (size_t)which * Ddim * Ddim;
    const float*   bias = (which == 0) ? bq : (which == 1) ? bk : bv;
    _Float16*      out  = (which == 0) ? qf : (which == 1) ? kf : vf;

    size_t row0 = (size_t)blockIdx.x * 16;
    int    col0 = ntile * 16;

    v8f acc = zero8();
    for (int k0 = 0; k0 < Ddim; k0 += 32) {
        __builtin_prefetch(H + row0 * Ddim + k0 + 64, 0, 0);  // global_prefetch_b8
        v16h a = load_a_f32(H, Ddim, row0, k0, lane);
        v16h b = load_b_f16t(Wt, Ddim, k0, col0, lane);
        acc = wmma_f16(a, b, acc);
    }
    int half = lane >> 4, n = lane & 15;
    float bn = bias[col0 + n];
#pragma unroll
    for (int r = 0; r < 8; ++r) {
        int M = r + half * 8;
        out[(row0 + M) * Ddim + col0 + n] = (_Float16)(acc[r] + bn);
    }
}

// ---------------------------------------------------------------------------
// Kernel 2: masked multi-head attention.  One wave per (b, h, 16-query tile).
// grid = B*NH*(T/16) = 12288, block = 32.
// Pipeline: TDM(K slab) -> Q.K^T -> TDM(V slab, reusing LDS) || softmax
//           -> attn.V.  LDS: 16.6KB scores + 32KB slab.
// ---------------------------------------------------------------------------
__global__ void attn_kernel(const _Float16* __restrict__ qf,
                            const _Float16* __restrict__ kf,
                            const _Float16* __restrict__ vf,
                            const int* __restrict__ num_turns,
                            _Float16* __restrict__ ctxf) {
    __shared__ float    sc[16][SCPITCH];          // score / attn tile
    __shared__ _Float16 slab[Tdim * DHdim];       // K slab, then V slab

    int lane = threadIdx.x & 31;
    int id = blockIdx.x;
    int qt = id & 15;      id >>= 4;
    int h  = id % NHdim;
    int b  = id / NHdim;

    int nturn = num_turns[b];
    int half = lane >> 4, m = lane & 15, n = m;
    size_t rowq0 = (size_t)b * Tdim + qt * 16;
    size_t kbase = (size_t)b * Tdim;
    int colh = h * DHdim;

    // --- TDM: stage K[b, :, h, :] (256 x 64 f16) into LDS ------------------
    tdm_load_2d_f16((unsigned)(uintptr_t)slab, kf + kbase * Ddim + colh,
                    /*tensor_w=*/Ddim, /*tensor_h=*/Bdim * Tdim,
                    /*tile_w=*/DHdim, /*tile_h=*/Tdim);

    // Q fragment (16 queries x 64 dims) while the TDM runs
    v16h aq0 = load_a_f16(qf, Ddim, rowq0, colh, lane);
    v16h aq1 = load_a_f16(qf, Ddim, rowq0, colh + 32, lane);

    __builtin_amdgcn_s_wait_tensorcnt(0);
    __syncthreads();

    // scores = Q K^T / sqrt(DH), masked, into LDS
    for (int kt = 0; kt < 16; ++kt) {
        // B-fragment: col N = key kt*16+n, 16 contiguous d-values per lane
        const _Float16* kp = slab + (kt * 16 + n) * DHdim + half * 16;
        v16h b0 = cat8(*(const v8h*)kp,        *(const v8h*)(kp + 8));
        v16h b1 = cat8(*(const v8h*)(kp + 32), *(const v8h*)(kp + 40));
        v8f s = zero8();
        s = wmma_f16(aq0, b0, s);
        s = wmma_f16(aq1, b1, s);
#pragma unroll
        for (int r = 0; r < 8; ++r) {
            int M   = r + half * 8;
            int tq  = qt * 16 + M;
            int key = kt * 16 + n;
            float val = s[r] * 0.125f;  // 1/sqrt(64)
            sc[M][key] = ((key < tq) && (key < nturn)) ? val : -1e30f;
        }
    }
    __syncthreads();   // K slab dead from here; scores visible

    // --- TDM: stage V[b, :, h, :] into the same LDS slab, overlapping the
    //     DMA with the softmax below.
    tdm_load_2d_f16((unsigned)(uintptr_t)slab, vf + kbase * Ddim + colh,
                    Ddim, Bdim * Tdim, DHdim, Tdim);

    // per-row softmax; row t==0 has no valid keys -> zero row (row_valid)
    if (lane < 16) {
        int tq = qt * 16 + lane;
        if (tq == 0) {
            for (int k = 0; k < Tdim; ++k) sc[lane][k] = 0.f;
        } else {
            float mx = -1e30f;
            for (int k = 0; k < Tdim; ++k) mx = fmaxf(mx, sc[lane][k]);
            float sum = 0.f;
            for (int k = 0; k < Tdim; ++k) {
                float p = __expf(sc[lane][k] - mx);
                sc[lane][k] = p;
                sum += p;
            }
            float inv = 1.f / sum;
            for (int k = 0; k < Tdim; ++k) sc[lane][k] *= inv;
        }
    }
    __builtin_amdgcn_s_wait_tensorcnt(0);
    __syncthreads();

    // ctx = attn @ V : 8 key-blocks of 32, 4 output d-tiles of 16
    v8f acc0 = zero8(), acc1 = zero8(), acc2 = zero8(), acc3 = zero8();
    for (int kb = 0; kb < 8; ++kb) {
        // A-fragment from LDS scores: contiguous 16B runs -> ds_load_b128 x4
        const float* sp = &sc[m][kb * 32 + half * 16];
        v4f s0 = *(const v4f*)sp;
        v4f s1 = *(const v4f*)(sp + 4);
        v4f s2 = *(const v4f*)(sp + 8);
        v4f s3 = *(const v4f*)(sp + 12);
        v16h a;
#pragma unroll
        for (int i = 0; i < 4; ++i) {
            a[i]      = (_Float16)s0[i];
            a[i + 4]  = (_Float16)s1[i];
            a[i + 8]  = (_Float16)s2[i];
            a[i + 12] = (_Float16)s3[i];
        }
#pragma unroll
        for (int ntile = 0; ntile < 4; ++ntile) {
            v16h bv;
#pragma unroll
            for (int j = 0; j < 8; ++j) {
                int kkey = kb * 32 + half * 16 + 2 * j;
                const _Float16* vp = slab + kkey * DHdim + ntile * 16 + n;
                bv[2 * j]     = vp[0];
                bv[2 * j + 1] = vp[DHdim];
            }
            if      (ntile == 0) acc0 = wmma_f16(a, bv, acc0);
            else if (ntile == 1) acc1 = wmma_f16(a, bv, acc1);
            else if (ntile == 2) acc2 = wmma_f16(a, bv, acc2);
            else                 acc3 = wmma_f16(a, bv, acc3);
        }
    }
#pragma unroll
    for (int r = 0; r < 8; ++r) {
        int M = r + half * 8;
        _Float16* cp = ctxf + (rowq0 + M) * Ddim + colh + n;
        cp[0]  = (_Float16)acc0[r];
        cp[16] = (_Float16)acc1[r];
        cp[32] = (_Float16)acc2[r];
        cp[48] = (_Float16)acc3[r];
    }
}

// ---------------------------------------------------------------------------
// Kernel 3: c = ctx @ Wo + bo, rows with t==0 forced to exact zero.
// grid=(1024,6), block=256 (8 waves).
// ---------------------------------------------------------------------------
__global__ void outproj_kernel(const _Float16* __restrict__ ctxf,
                               const _Float16* __restrict__ woT,  // f16 [N][K]
                               const float* __restrict__ bo,
                               _Float16* __restrict__ cf) {
    int lane = threadIdx.x & 31;
    int w    = threadIdx.x >> 5;
    int ntile = blockIdx.y * 8 + w;        // 0..47
    size_t row0 = (size_t)blockIdx.x * 16;
    int    col0 = ntile * 16;

    v8f acc = zero8();
    for (int k0 = 0; k0 < Ddim; k0 += 32) {
        __builtin_prefetch(ctxf + row0 * Ddim + k0 + 64, 0, 0);
        v16h a = load_a_f16(ctxf, Ddim, row0, k0, lane);
        v16h b = load_b_f16t(woT, Ddim, k0, col0, lane);
        acc = wmma_f16(a, b, acc);
    }
    int half = lane >> 4, n = lane & 15;
    float bn = bo[col0 + n];
#pragma unroll
    for (int r = 0; r < 8; ++r) {
        int M = r + half * 8;
        size_t row = row0 + M;
        int t = (int)(row & (Tdim - 1));
        float val = acc[r] + bn;
        if (t == 0) val = 0.f;             // row_valid == false
        cf[row * Ddim + col0 + n] = (_Float16)val;
    }
}

// ---------------------------------------------------------------------------
// Kernel 4: all_logits = [H, c] @ Wc + bc.  K=1536, N=7 padded to 16.
// grid=128, block=256 (8 waves).
// ---------------------------------------------------------------------------
__global__ void cls_kernel(const float* __restrict__ H,
                           const _Float16* __restrict__ cf,
                           const float* __restrict__ Wc,
                           const float* __restrict__ bc,
                           float* __restrict__ logits) {
    int lane = threadIdx.x & 31;
    int w    = threadIdx.x >> 5;
    size_t row0 = ((size_t)blockIdx.x * 8 + w) * 16;

    int half = lane >> 4, n = lane & 15;
    v8f acc = zero8();
    // K = 0..767: A from H (fp32, wide loads)
    for (int k0 = 0; k0 < Ddim; k0 += 32) {
        v16h a = load_a_f32(H, Ddim, row0, k0, lane);
        v16h b;
#pragma unroll
        for (int j = 0; j < 8; ++j) {
            int k = k0 + half * 16 + 2 * j;
            b[2 * j]     = (n < Cdim) ? (_Float16)Wc[(size_t)k * Cdim + n]
                                      : (_Float16)0.f;
            b[2 * j + 1] = (n < Cdim) ? (_Float16)Wc[(size_t)(k + 1) * Cdim + n]
                                      : (_Float16)0.f;
        }
        acc = wmma_f16(a, b, acc);
    }
    // K = 768..1535: A from c (f16, wide loads)
    for (int k0 = 0; k0 < Ddim; k0 += 32) {
        v16h a = load_a_f16(cf, Ddim, row0, k0, lane);
        v16h b;
#pragma unroll
        for (int j = 0; j < 8; ++j) {
            int k = Ddim + k0 + half * 16 + 2 * j;
            b[2 * j]     = (n < Cdim) ? (_Float16)Wc[(size_t)k * Cdim + n]
                                      : (_Float16)0.f;
            b[2 * j + 1] = (n < Cdim) ? (_Float16)Wc[(size_t)(k + 1) * Cdim + n]
                                      : (_Float16)0.f;
        }
        acc = wmma_f16(a, b, acc);
    }
    if (n < Cdim) {
        float bn = bc[n];
#pragma unroll
        for (int r = 0; r < 8; ++r) {
            int M = r + half * 8;
            logits[(row0 + M) * Cdim + n] = acc[r] + bn;
        }
    }
}

// ---------------------------------------------------------------------------
// Kernel 5: final_logits[b] = all_logits[b, num_turns[b]-1]
// ---------------------------------------------------------------------------
__global__ void final_kernel(const float* __restrict__ logits,
                             const int* __restrict__ num_turns,
                             float* __restrict__ fin) {
    int idx = blockIdx.x * blockDim.x + threadIdx.x;
    if (idx < Bdim * Cdim) {
        int b = idx / Cdim, c = idx % Cdim;
        int t = num_turns[b] - 1;
        fin[idx] = logits[((size_t)b * Tdim + t) * Cdim + c];
    }
}

// ---------------------------------------------------------------------------
extern "C" void kernel_launch(void* const* d_in, const int* in_sizes, int n_in,
                              void* d_out, int out_size, void* d_ws, size_t ws_size,
                              hipStream_t stream) {
    (void)in_sizes; (void)n_in; (void)out_size; (void)ws_size;
    const float* H  = (const float*)d_in[0];
    const float* Wq = (const float*)d_in[1];
    const float* bq = (const float*)d_in[2];
    const float* Wk = (const float*)d_in[3];
    const float* bk = (const float*)d_in[4];
    const float* Wv = (const float*)d_in[5];
    const float* bv = (const float*)d_in[6];
    const float* Wo = (const float*)d_in[7];
    const float* bo = (const float*)d_in[8];
    const float* Wc = (const float*)d_in[9];
    const float* bc = (const float*)d_in[10];
    const int*   nt = (const int*)d_in[11];

    float*     out = (float*)d_out;
    _Float16*  ws  = (_Float16*)d_ws;
    const size_t MAT  = (size_t)Bdim * Tdim * Ddim;  // 12.58M f16 elems
    const size_t WMAT = (size_t)Ddim * Ddim;         // 589.8K f16 elems
    _Float16* qf   = ws;
    _Float16* kf   = ws + MAT;
    _Float16* vf   = ws + 2 * MAT;
    _Float16* ctxf = ws + 3 * MAT;
    _Float16* cf   = ws + 4 * MAT;
    _Float16* wt   = ws + 5 * MAT;                   // 4 packed weight mats

    wpack_kernel<<<dim3((unsigned)((4 * WMAT + 255) / 256)), 256, 0, stream>>>(
        Wq, Wk, Wv, Wo, wt);
    qkv_kernel<<<dim3(1024, 18), 256, 0, stream>>>(H, wt, bq, bk, bv,
                                                   qf, kf, vf);
    attn_kernel<<<dim3(Bdim * NHdim * (Tdim / 16)), 32, 0, stream>>>(qf, kf, vf,
                                                                     nt, ctxf);
    outproj_kernel<<<dim3(1024, 6), 256, 0, stream>>>(ctxf, wt + 3 * WMAT, bo, cf);
    cls_kernel<<<dim3(128), 256, 0, stream>>>(H, cf, Wc, bc, out);
    final_kernel<<<dim3(2), 256, 0, stream>>>(out, nt,
                                              out + (size_t)Bdim * Tdim * Cdim);
}
